// MC_Net_43954695307534
// MI455X (gfx1250) — matrix operand order repaired and checked by reading, not compiled, not run on hardware
//
#include <hip/hip_runtime.h>
#include <hip/hip_bf16.h>

typedef __attribute__((ext_vector_type(16))) _Float16 v16h;
typedef __attribute__((ext_vector_type(8)))  _Float16 v8h;
typedef __attribute__((ext_vector_type(8)))  float    v8f;

#define BB 4
#define HH 192
#define WW 192
#define NPIX (BB * HH * WW)        // 147456
#define TILES_X (WW / 16)          // 12
#define NTILE (BB * HH * TILES_X)  // 9216  (deform: 16-px tiles)
#define TILES_X2 (WW / 32)         // 6
#define NTILE2 (BB * HH * TILES_X2) // 4608 (conv: 32-px double tiles)

// ---------------------------------------------------------------------------
// Layout transforms
// ---------------------------------------------------------------------------

// NCHW f32 -> NHWC f16   (pixel index p within image, channel contiguous)
__global__ void pack_nhwc_f16(const float* __restrict__ in,
                              _Float16* __restrict__ out, int C, int total) {
  int idx = blockIdx.x * 256 + threadIdx.x;
  if (idx >= total) return;
  int p  = idx % (HH * WW);
  int bc = idx / (HH * WW);
  int c  = bc % C;
  int b  = bc / C;
  out[((size_t)b * (HH * WW) + p) * C + c] = (_Float16)in[idx];
}

// OIHW f32 conv weights -> WMMA B-fragment pack
// out[((kp*nChunk + chunk)*nCot + cot)*512 + lane*16 + i]
//   lane L, half i:  N(co) = cot*16 + L%16 ; K(ci) = chunk*32 + 8*(L/16) + (i%8) + 16*(i/8)
__global__ void pack_wconv(const float* __restrict__ w,
                           _Float16* __restrict__ out, int Cin, int Cout) {
  int nChunk = Cin / 32, nCot = Cout / 16;
  int total = 9 * nChunk * nCot * 512;
  int idx = blockIdx.x * 256 + threadIdx.x;
  if (idx >= total) return;
  int i    = idx & 15;
  int lane = (idx >> 4) & 31;
  int rest = idx >> 9;
  int cot   = rest % nCot;
  int chunk = (rest / nCot) % nChunk;
  int kp    = rest / (nCot * nChunk);
  int co = cot * 16 + (lane & 15);
  int ci = chunk * 32 + 8 * (lane >> 4) + (i & 7) + 16 * (i >> 3);
  int ky = kp / 3, kx = kp % 3;
  out[idx] = (_Float16)w[((size_t)(co * Cin + ci) * 3 + ky) * 3 + kx];
}

// DCN weights [64,64,3,3] -> B pack over K_gemm = kp*64 + c  (18 chunks of 32)
// out[(chunkG*4 + cot)*512 + lane*16 + i]
__global__ void pack_wdcn(const float* __restrict__ w, _Float16* __restrict__ out) {
  int total = 18 * 4 * 512;
  int idx = blockIdx.x * 256 + threadIdx.x;
  if (idx >= total) return;
  int i    = idx & 15;
  int lane = (idx >> 4) & 31;
  int rest = idx >> 9;
  int cot    = rest & 3;
  int chunkG = rest >> 2;
  int kg = chunkG * 32 + 8 * (lane >> 4) + (i & 7) + 16 * (i >> 3);   // K_gemm
  int c  = kg & 63;                               // input channel (g*8+ci)
  int kp = kg >> 6;                               // kernel tap
  int oc = cot * 16 + (lane & 15);
  out[idx] = (_Float16)w[((size_t)oc * 64 + c) * 9 + kp];
}

// ---------------------------------------------------------------------------
// Implicit-GEMM 3x3 conv (pad=1) via v_wmma_f32_16x16x32_f16
// One wave per (32 pixels in a row = two 16-px N-tiles) x (16 output channels).
// Each B (weight) fragment is loaded once and reused for both N-tiles.
// ---------------------------------------------------------------------------
template <int CIN, bool RELU, bool STOREF32>
__global__ void __launch_bounds__(32)
conv3x3_wmma(const _Float16* __restrict__ in, const _Float16* __restrict__ wpack,
             const float* __restrict__ bias, void* __restrict__ outp,
             int out_cstride, int co_off) {
  constexpr int NCHUNK = CIN / 32;
  const int nCot = gridDim.y;
  const int cot  = blockIdx.y;
  int t = blockIdx.x;
  const int xt = t % TILES_X2; t /= TILES_X2;
  const int y0 = t % HH;
  const int b  = t / HH;
  const int lane = threadIdx.x;
  const int m = lane & 15, h = lane >> 4;
  const int px0 = xt * 32 + m;        // N-tile 0
  const int px1 = px0 + 16;           // N-tile 1

  const _Float16* inb = in + (size_t)b * (HH * WW) * CIN;
  v8f acc0 = {};
  v8f acc1 = {};

  for (int ky = 0; ky < 3; ++ky) {
    const int py = y0 + ky - 1;
    const bool yok = (py >= 0) && (py < HH);
    const size_t rowoff = (size_t)(yok ? py : 0) * WW;
#pragma unroll
    for (int kx = 0; kx < 3; ++kx) {
      const int qx0 = px0 + kx - 1;
      const int qx1 = px1 + kx - 1;
      const bool ok0 = yok && (qx0 >= 0) && (qx0 < WW);
      const bool ok1 = yok && (qx1 >= 0) && (qx1 < WW);
      const _Float16* src0 = inb + (rowoff + (ok0 ? qx0 : 0)) * CIN;
      const _Float16* src1 = inb + (rowoff + (ok1 ? qx1 : 0)) * CIN;
      const int kp = ky * 3 + kx;
      const _Float16* wb = wpack + ((size_t)(kp * NCHUNK) * nCot + cot) * 512;
      __builtin_prefetch(wb + (size_t)NCHUNK * nCot * 512, 0, 0);
#pragma unroll
      for (int ch = 0; ch < NCHUNK; ++ch) {
        v16h bf = *(const v16h*)(wb + (size_t)ch * nCot * 512 + lane * 16);
        v16h a0 = {};
        if (ok0) {
          v8h lo = *(const v8h*)(src0 + ch * 32 + 8 * h);
          v8h hi = *(const v8h*)(src0 + ch * 32 + 16 + 8 * h);
#pragma unroll
          for (int i = 0; i < 8; ++i) { a0[i] = lo[i]; a0[i + 8] = hi[i]; }
        }
        v16h a1 = {};
        if (ok1) {
          v8h lo = *(const v8h*)(src1 + ch * 32 + 8 * h);
          v8h hi = *(const v8h*)(src1 + ch * 32 + 16 + 8 * h);
#pragma unroll
          for (int i = 0; i < 8; ++i) { a1[i] = lo[i]; a1[i + 8] = hi[i]; }
        }
        acc0 = __builtin_amdgcn_wmma_f32_16x16x32_f16(false, a0, false, bf,
                                                      (short)0, acc0, false, false);
        acc1 = __builtin_amdgcn_wmma_f32_16x16x32_f16(false, a1, false, bf,
                                                      (short)0, acc1, false, false);
      }
    }
  }

  const int co = cot * 16 + m;
  const float bv = bias[co];
  const size_t pixbase = (size_t)b * (HH * WW) + (size_t)y0 * WW + xt * 32;
#pragma unroll
  for (int g = 0; g < 8; ++g) {
    float v0 = acc0[g] + bv;
    float v1 = acc1[g] + bv;
    if (RELU) {
      v0 = v0 > 0.f ? v0 : 0.f;
      v1 = v1 > 0.f ? v1 : 0.f;
    }
    const size_t pix0 = pixbase + g + 8 * h;
    const size_t pix1 = pix0 + 16;
    if (STOREF32) {
      ((float*)outp)[pix0 * out_cstride + co_off + co] = v0;
      ((float*)outp)[pix1 * out_cstride + co_off + co] = v1;
    } else {
      ((_Float16*)outp)[pix0 * out_cstride + co_off + co] = (_Float16)v0;
      ((_Float16*)outp)[pix1 * out_cstride + co_off + co] = (_Float16)v1;
    }
  }
}

// ---------------------------------------------------------------------------
// Deformable conv: bilinear-sample 16px x 576(K) into LDS, then WMMA GEMM
// ---------------------------------------------------------------------------
__global__ void __launch_bounds__(32)
deform_wmma(const _Float16* __restrict__ ref, const float* __restrict__ off,
            const _Float16* __restrict__ wpack, const float* __restrict__ bias,
            _Float16* __restrict__ out, int out_cstride, int co_off) {
  __shared__ __align__(16) _Float16 smem[16 * 576];
  int t = blockIdx.x;
  const int xt = t % TILES_X; t /= TILES_X;
  const int y0 = t % HH;
  const int b  = t / HH;
  const int lane = threadIdx.x;
  const _Float16* refb = ref + (size_t)b * (HH * WW) * 64;
  const size_t pixbase = (size_t)b * (HH * WW) + (size_t)y0 * WW + xt * 16;

  // --- sampling: 16 pixels x 8 groups x 9 taps, 8 channels per group ---
  for (int s = lane; s < 16 * 72; s += 32) {
    const int m = s / 72;
    const int j = s % 72;
    const int g = j / 9, kp = j % 9;
    const int ky = kp / 3, kx = kp % 3;
    const float* op = off + (pixbase + m) * 144 + (size_t)(g * 9 + kp) * 2;
    const float dy = op[0], dx = op[1];
    const float pyf = (float)(y0 + ky - 1) + dy;
    const float pxf = (float)(xt * 16 + m + kx - 1) + dx;
    const float fy = floorf(pyf), fx = floorf(pxf);
    const float wy = pyf - fy, wx = pxf - fx;
    const int yi = (int)fy, xi = (int)fx;
    const float w00 = (1.f - wy) * (1.f - wx), w01 = (1.f - wy) * wx;
    const float w10 = wy * (1.f - wx), w11 = wy * wx;

    auto tap = [&](int yc, int xc, float& wv) -> const _Float16* {
      bool valid = (yc >= 0) && (yc < HH) && (xc >= 0) && (xc < WW);
      if (!valid) wv = 0.f;
      int yy = yc < 0 ? 0 : (yc > HH - 1 ? HH - 1 : yc);
      int xx = xc < 0 ? 0 : (xc > WW - 1 ? WW - 1 : xc);
      return refb + ((size_t)yy * WW + xx) * 64 + g * 8;
    };
    float a00 = w00, a01 = w01, a10 = w10, a11 = w11;
    const _Float16* p00 = tap(yi,     xi,     a00);
    const _Float16* p01 = tap(yi,     xi + 1, a01);
    const _Float16* p10 = tap(yi + 1, xi,     a10);
    const _Float16* p11 = tap(yi + 1, xi + 1, a11);
#pragma unroll
    for (int ci = 0; ci < 8; ++ci) {
      float v = a00 * (float)p00[ci] + a01 * (float)p01[ci] +
                a10 * (float)p10[ci] + a11 * (float)p11[ci];
      smem[m * 576 + kp * 64 + g * 8 + ci] = (_Float16)v;  // K = kp*64 + c
    }
  }
  __syncthreads();

  // --- GEMM: 16px x 576K x 64co ---
  const int m = lane & 15, h = lane >> 4;
  for (int cot = 0; cot < 4; ++cot) {
    v8f acc = {};
#pragma unroll
    for (int ch = 0; ch < 18; ++ch) {
      const _Float16* sp = smem + m * 576 + ch * 32;
      v8h lo = *(const v8h*)(sp + 8 * h);
      v8h hi = *(const v8h*)(sp + 16 + 8 * h);
      v16h a;
#pragma unroll
      for (int i = 0; i < 8; ++i) { a[i] = lo[i]; a[i + 8] = hi[i]; }
      v16h bf = *(const v16h*)(wpack + ((size_t)ch * 4 + cot) * 512 + lane * 16);
      acc = __builtin_amdgcn_wmma_f32_16x16x32_f16(false, a, false, bf,
                                                   (short)0, acc, false, false);
    }
    const int co = cot * 16 + m;
    const float bv = bias[co];
#pragma unroll
    for (int g = 0; g < 8; ++g) {
      float v = acc[g] + bv;
      v = v > 0.f ? v : 0.f;  // relu(def_feat)
      out[(pixbase + g + 8 * h) * out_cstride + co_off + co] = (_Float16)v;
    }
  }
}

// ---------------------------------------------------------------------------
// SE block + final combine
// ---------------------------------------------------------------------------
__global__ void reduce_mean(const _Float16* __restrict__ x2,
                            float* __restrict__ g) {
  const int b  = blockIdx.x >> 6;
  const int co = blockIdx.x & 63;
  const _Float16* p = x2 + (size_t)b * (HH * WW) * 64 + co;
  float s = 0.f;
  for (int i = threadIdx.x; i < HH * WW; i += 256) s += (float)p[(size_t)i * 64];
  __shared__ float red[256];
  red[threadIdx.x] = s;
  __syncthreads();
  for (int st = 128; st > 0; st >>= 1) {
    if (threadIdx.x < st) red[threadIdx.x] += red[threadIdx.x + st];
    __syncthreads();
  }
  if (threadIdx.x == 0) g[blockIdx.x] = red[0] / (float)(HH * WW);
}

__global__ void se_kernel(const float* __restrict__ g,
                          const float* __restrict__ w1, const float* __restrict__ b1,
                          const float* __restrict__ w2, const float* __restrict__ b2,
                          float* __restrict__ a) {
  __shared__ float hid[4 * 32];
  const int t = threadIdx.x;
  if (t < 128) {
    const int b = t >> 5, hh = t & 31;
    float s = b1[hh];
    for (int c = 0; c < 64; ++c) s += g[b * 64 + c] * w1[hh * 64 + c];
    hid[b * 32 + hh] = s > 0.f ? s : 0.f;
  }
  __syncthreads();
  {
    const int b = t >> 6, co = t & 63;
    float s = b2[co];
    for (int k = 0; k < 32; ++k) s += hid[b * 32 + k] * w2[co * 32 + k];
    a[b * 64 + co] = 1.f / (1.f + expf(-s));
  }
}

__global__ void final_combine(const _Float16* __restrict__ cat,
                              const _Float16* __restrict__ x2,
                              const float* __restrict__ a,
                              float* __restrict__ out) {
  size_t idx = (size_t)blockIdx.x * 256 + threadIdx.x;
  if (idx >= (size_t)BB * 64 * HH * WW) return;
  const int p  = (int)(idx % (HH * WW));
  const int co = (int)((idx / (HH * WW)) % 64);
  const int b  = (int)(idx / ((size_t)64 * HH * WW));
  const size_t pix = (size_t)b * (HH * WW) + p;
  const float def = (float)cat[pix * 128 + co];
  const float xv  = (float)x2[pix * 64 + co];
  out[idx] = def + xv * a[b * 64 + co];
}

// ---------------------------------------------------------------------------
// Host orchestration
// ---------------------------------------------------------------------------
extern "C" void kernel_launch(void* const* d_in, const int* in_sizes, int n_in,
                              void* d_out, int out_size, void* d_ws, size_t ws_size,
                              hipStream_t stream) {
  (void)in_sizes; (void)n_in; (void)out_size; (void)ws_size;
  const float* mv     = (const float*)d_in[0];
  const float* ref    = (const float*)d_in[1];
  const float* head_w = (const float*)d_in[2];
  const float* head_b = (const float*)d_in[3];
  const float* dcn_w  = (const float*)d_in[4];
  const float* dcn_b  = (const float*)d_in[5];
  const float* c1w1   = (const float*)d_in[6];
  const float* c1b1   = (const float*)d_in[7];
  const float* c1w2   = (const float*)d_in[8];
  const float* c1b2   = (const float*)d_in[9];
  const float* aux_w  = (const float*)d_in[10];
  const float* aux_b  = (const float*)d_in[11];
  const float* fw1    = (const float*)d_in[12];
  const float* fb1    = (const float*)d_in[13];
  const float* fw2    = (const float*)d_in[14];
  const float* fb2    = (const float*)d_in[15];
  const float* se_w1  = (const float*)d_in[16];
  const float* se_b1  = (const float*)d_in[17];
  const float* se_w2  = (const float*)d_in[18];
  const float* se_b2  = (const float*)d_in[19];

  char* ws = (char*)d_ws;
  size_t cur = 0;
  auto alloc = [&](size_t bytes) -> char* {
    size_t r = (cur + 255) & ~(size_t)255;
    cur = r + bytes;
    return ws + r;
  };

  _Float16* mv_nhwc  = (_Float16*)alloc((size_t)NPIX * 128 * 2); // reused as aux_in
  _Float16* ref_nhwc = (_Float16*)alloc((size_t)NPIX * 64 * 2);
  float*    off_nhwc = (float*)   alloc((size_t)NPIX * 144 * 4);
  _Float16* c1a      = (_Float16*)alloc((size_t)NPIX * 128 * 2); // reused as x1
  _Float16* catb     = (_Float16*)alloc((size_t)NPIX * 128 * 2); // [def_feat | aux]
  _Float16* x2       = (_Float16*)alloc((size_t)NPIX * 64 * 2);
  _Float16* pk_head  = (_Float16*)alloc((size_t)128 * 144 * 9 * 2);
  _Float16* pk_c1w1  = (_Float16*)alloc((size_t)64 * 128 * 9 * 2);
  _Float16* pk_c1w2  = (_Float16*)alloc((size_t)128 * 128 * 9 * 2);
  _Float16* pk_aux   = (_Float16*)alloc((size_t)128 * 64 * 9 * 2);
  _Float16* pk_fw1   = (_Float16*)alloc((size_t)128 * 64 * 9 * 2);
  _Float16* pk_fw2   = (_Float16*)alloc((size_t)64 * 64 * 9 * 2);
  _Float16* pk_dcn   = (_Float16*)alloc((size_t)64 * 64 * 9 * 2);
  float*    gbuf     = (float*)alloc(256 * 4);
  float*    abuf     = (float*)alloc(256 * 4);

  // 1) activation repack
  {
    int tot = BB * 128 * HH * WW;
    pack_nhwc_f16<<<(tot + 255) / 256, 256, 0, stream>>>(mv, mv_nhwc, 128, tot);
  }
  {
    int tot = BB * 64 * HH * WW;
    pack_nhwc_f16<<<(tot + 255) / 256, 256, 0, stream>>>(ref, ref_nhwc, 64, tot);
  }

  // 2) weight repack
  auto packW = [&](const float* w, _Float16* out, int Cin, int Cout) {
    int tot = 9 * Cin * Cout;
    pack_wconv<<<(tot + 255) / 256, 256, 0, stream>>>(w, out, Cin, Cout);
  };
  packW(head_w, pk_head, 128, 144);
  packW(c1w1,   pk_c1w1, 64, 128);
  packW(c1w2,   pk_c1w2, 128, 128);
  packW(aux_w,  pk_aux,  128, 64);
  packW(fw1,    pk_fw1,  128, 64);
  packW(fw2,    pk_fw2,  64, 64);
  pack_wdcn<<<(18 * 4 * 512 + 255) / 256, 256, 0, stream>>>(dcn_w, pk_dcn);

  // 3) offsets = conv(mv, head_w)  -> f32 NHWC (needed at full precision)
  conv3x3_wmma<128, false, true><<<dim3(NTILE2, 9), 32, 0, stream>>>(
      mv_nhwc, pk_head, head_b, off_nhwc, 144, 0);

  // 4) c1: relu(conv(ref, c1w1)) -> c1a ; relu(conv(c1a, c1w2)) -> aux_in (mv buf)
  conv3x3_wmma<64, true, false><<<dim3(NTILE2, 8), 32, 0, stream>>>(
      ref_nhwc, pk_c1w1, c1b1, c1a, 128, 0);
  conv3x3_wmma<128, true, false><<<dim3(NTILE2, 8), 32, 0, stream>>>(
      c1a, pk_c1w2, c1b2, mv_nhwc, 128, 0);

  // 5) aux = relu(conv(aux_in, aux_w)) -> cat[:,64:128]
  conv3x3_wmma<128, true, false><<<dim3(NTILE2, 4), 32, 0, stream>>>(
      mv_nhwc, pk_aux, aux_b, catb, 128, 64);

  // 6) def_feat = relu(deform_conv(ref, offsets, dcn_w)) -> cat[:,0:64]
  deform_wmma<<<NTILE, 32, 0, stream>>>(ref_nhwc, off_nhwc, pk_dcn, dcn_b,
                                        catb, 128, 0);

  // 7) x = relu(conv(cat, fw1)) -> x1 (c1a buf) ; x = conv(x1, fw2) -> x2
  conv3x3_wmma<128, true, false><<<dim3(NTILE2, 4), 32, 0, stream>>>(
      catb, pk_fw1, fb1, c1a, 64, 0);
  conv3x3_wmma<64, false, false><<<dim3(NTILE2, 4), 32, 0, stream>>>(
      c1a, pk_fw2, fb2, x2, 64, 0);

  // 8) SE: g = mean(x2) ; a = sigmoid(relu(g@w1.T+b1)@w2.T+b2)
  reduce_mean<<<BB * 64, 256, 0, stream>>>(x2, gbuf);
  se_kernel<<<1, 256, 0, stream>>>(gbuf, se_w1, se_b1, se_w2, se_b2, abuf);

  // 9) out = def_feat + x2 * a   (NCHW f32)
  {
    int tot = BB * 64 * HH * WW;
    final_combine<<<(tot + 255) / 256, 256, 0, stream>>>(catb, x2, abuf,
                                                         (float*)d_out);
  }
}
